// IGNN_81664508166412
// MI455X (gfx1250) — matrix operand order repaired and checked by reading, not compiled, not run on hardware
//
#include <hip/hip_runtime.h>
#include <hip/hip_bf16.h>

// ---------------------------------------------------------------------------
// Interaction-network GNN step for MI455X (gfx1250, wave32, WMMA).
// GEMMs: v_wmma_f32_16x16x32_bf16, fp32 accumulate.
// Block = 64 rows (4 M-tiles): each B-fragment load feeds 4 WMMAs (4x weight
// reuse, and the 4-deep accumulator interleave covers the WMMA->WMMA hazard).
// Staging is float4-vectorized: global_load_b128 -> packed bf16 -> ds_store_b64.
// Node table (25.6MB) + agg buffer are L2-resident; HBM traffic = edge streams.
// ---------------------------------------------------------------------------

typedef __attribute__((ext_vector_type(16))) __bf16 v16bf;
typedef __attribute__((ext_vector_type(8)))  float  v8f;

#define D_   128   // node feature dim
#define DE_  128   // edge feature dim
#define H_   256   // hidden dim
#define XIN_ 384   // DE + 2D
#define MT_  4     // M-tiles per block (64 rows)

__device__ __forceinline__ v16bf load_frag_pair(const __bf16* lo, const __bf16* hi) {
    union { v16bf v; uint4 q[2]; } u;
    u.q[0] = *(const uint4*)lo;
    u.q[1] = *(const uint4*)hi;
    return u.v;
}

// Convert 4 contiguous f32 to bf16 and store 8 bytes into LDS.
__device__ __forceinline__ void cvt_store4(__bf16* dst, const float4 v) {
    union { __bf16 h[4]; uint2 u; } t;
    t.h[0] = (__bf16)v.x; t.h[1] = (__bf16)v.y;
    t.h[2] = (__bf16)v.z; t.h[3] = (__bf16)v.w;
    *(uint2*)dst = t.u;
}

// 4 M-tiles x 1 N-tile of  C = X[64 x 32*ktiles] @ W  (B-fragment reused 4x).
//   X  : LDS bf16 row-major, leading dim ldx (multiple of 8).
//        A-frag (16-bit, 16x32): lanes 0-15 hold K {0..7,16..23} of row M=lane,
//        lanes 16-31 hold K {8..15,24..31} of row M=lane-16.
//   Wp : global bf16, pre-packed per-lane-contiguous:
//        frag(kt,nt) at ((kt*ntiles+nt)*32 + lane)*16 ;  lane -> col n0+(lane&15),
//        element e -> K = kt*32 + (lane>=16?8:0) + (e&7) + (e>=8?16:0).
__device__ __forceinline__ void gemm_tile_m4(const __bf16* X, int ldx, int ktiles,
                                             const __bf16* Wp, int nt, int ntiles,
                                             int lane, v8f c[MT_])
{
    const int row = lane & 15;
    const int kb  = (lane & 16) ? 8 : 0;
    const __bf16* xbase = X + row * ldx + kb;
#pragma unroll
    for (int kt = 0; kt < ktiles; ++kt) {
        const __bf16* bp = Wp + (((size_t)kt * ntiles + nt) * 32 + (size_t)lane) * 16;
        v16bf b = load_frag_pair(bp, bp + 8);
#pragma unroll
        for (int m = 0; m < MT_; ++m) {
            const __bf16* xr = xbase + (m * 16) * ldx + kt * 32;
            v16bf a = load_frag_pair(xr, xr + 16);
            c[m] = __builtin_amdgcn_wmma_f32_16x16x32_bf16(false, a, false, b,
                                                           (short)0, c[m], false, false);
        }
    }
}

// ---------------------------------------------------------------------------
// Weight pre-pack: fp32 [K x Nw] row-major -> bf16 B-fragment layout.
// ---------------------------------------------------------------------------
__global__ void pack_w_kernel(const float* __restrict__ W, __bf16* __restrict__ out,
                              int K, int Nw)
{
    const int total  = K * Nw;
    const int ntiles = Nw >> 4;
    for (int p = blockIdx.x * blockDim.x + threadIdx.x; p < total;
         p += gridDim.x * blockDim.x) {
        int e    = p & 15;
        int lane = (p >> 4) & 31;
        int t    = p >> 9;
        int nt   = t % ntiles;
        int kt   = t / ntiles;
        int col  = nt * 16 + (lane & 15);
        int krow = kt * 32 + ((lane & 16) ? 8 : 0) + (e & 7) + ((e & 8) ? 16 : 0);
        out[p] = (__bf16)W[(size_t)krow * Nw + col];
    }
}

__global__ void zero_f32_kernel(float* __restrict__ p, int n)
{
    int i = blockIdx.x * blockDim.x + threadIdx.x;
    if (i < n) p[i] = 0.f;
}

// ---------------------------------------------------------------------------
// Fused edge kernel: one block = 64 edges, 128 threads = 4 waves.
//   xin = [edges | h_recv | h_send] (bf16 LDS)
//   hid = relu(xin @ We1 + be1)
//   edges_new = (hid @ We2 + be2) * active     -> d_out, and xin[:,0:128]
//   hid = relu(xin @ Wf1 + bf1)
//   effects = hid @ Wf2 + bf2                  -> atomicAdd agg[recv] (L2)
// ---------------------------------------------------------------------------
__global__ __launch_bounds__(128) void ign_edge_kernel(
    const float* __restrict__ nodes, const float* __restrict__ edges,
    const int* __restrict__ senders, const int* __restrict__ receivers,
    const float* __restrict__ active_edges,
    const __bf16* __restrict__ We1p, const float* __restrict__ be1,
    const __bf16* __restrict__ We2p, const float* __restrict__ be2,
    const __bf16* __restrict__ Wf1p, const float* __restrict__ bf1,
    const __bf16* __restrict__ Wf2p, const float* __restrict__ bf2,
    float* __restrict__ agg, float* __restrict__ edges_out, int E)
{
    __shared__ __align__(16) __bf16 xin[64][XIN_];   // 48 KB
    __shared__ __align__(16) __bf16 hid[64][H_];     // 32 KB
    __shared__ int s_recv[64];
    __shared__ int s_send[64];

    const int tid  = threadIdx.x;
    const int lane = tid & 31;
    const int wave = tid >> 5;
    const int e0   = blockIdx.x * 64;

    if (tid < 64) {
        int e = e0 + tid; if (e >= E) e = E - 1;
        s_recv[tid] = receivers[e];
        s_send[tid] = senders[e];
    }
    __syncthreads();

    // Vectorized staging: 64 rows x 128 cols per segment, float4 granularity.
    // i in [0, 64*32): row = i>>5, col4 = (i&31)*4.
#pragma unroll 4
    for (int i = tid; i < 64 * 32; i += 128) {
        const int row = i >> 5;
        const int c4  = (i & 31) << 2;
        int e  = e0 + row; if (e >= E) e = E - 1;
        const float4 ve = *(const float4*)(edges + (size_t)e * DE_ + c4);
        const float4 vr = *(const float4*)(nodes + (size_t)s_recv[row] * D_ + c4);
        const float4 vs = *(const float4*)(nodes + (size_t)s_send[row] * D_ + c4);
        cvt_store4(&xin[row][c4],       ve);
        cvt_store4(&xin[row][128 + c4], vr);
        cvt_store4(&xin[row][256 + c4], vs);
    }
    __syncthreads();

    const int mhi = (lane & 16) ? 8 : 0;
    const int nc  = lane & 15;

    // ---- edge MLP layer 1: [64,384] @ [384,256] ----
#pragma unroll 1
    for (int t = 0; t < 4; ++t) {
        int nt = wave * 4 + t;
        v8f c[MT_];
#pragma unroll
        for (int m = 0; m < MT_; ++m) c[m] = (v8f){0.f,0.f,0.f,0.f,0.f,0.f,0.f,0.f};
        gemm_tile_m4(&xin[0][0], XIN_, 12, We1p, nt, 16, lane, c);
        int N = nt * 16 + nc;
        float bias = be1[N];
#pragma unroll
        for (int m = 0; m < MT_; ++m)
#pragma unroll
            for (int r = 0; r < 8; ++r) {
                float h = c[m][r] + bias;
                hid[m * 16 + mhi + r][N] = (__bf16)(h > 0.f ? h : 0.f);
            }
    }
    __syncthreads();

    // ---- edge MLP layer 2: [64,256] @ [256,128] -> edges_new ----
#pragma unroll 1
    for (int t = 0; t < 2; ++t) {
        int nt = wave * 2 + t;
        v8f c[MT_];
#pragma unroll
        for (int m = 0; m < MT_; ++m) c[m] = (v8f){0.f,0.f,0.f,0.f,0.f,0.f,0.f,0.f};
        gemm_tile_m4(&hid[0][0], H_, 8, We2p, nt, 8, lane, c);
        int N = nt * 16 + nc;
        float bias = be2[N];
#pragma unroll
        for (int m = 0; m < MT_; ++m)
#pragma unroll
            for (int r = 0; r < 8; ++r) {
                int ridx = m * 16 + mhi + r;
                int e    = e0 + ridx;
                float ae = active_edges[(e < E) ? e : (E - 1)];
                float v  = (c[m][r] + bias) * ae;
                if (e < E) edges_out[(size_t)e * DE_ + N] = v;
                xin[ridx][N] = (__bf16)v;   // f_in = [edges_new | h_recv | h_send]
            }
    }
    __syncthreads();

    // ---- effect MLP layer 1 ----
#pragma unroll 1
    for (int t = 0; t < 4; ++t) {
        int nt = wave * 4 + t;
        v8f c[MT_];
#pragma unroll
        for (int m = 0; m < MT_; ++m) c[m] = (v8f){0.f,0.f,0.f,0.f,0.f,0.f,0.f,0.f};
        gemm_tile_m4(&xin[0][0], XIN_, 12, Wf1p, nt, 16, lane, c);
        int N = nt * 16 + nc;
        float bias = bf1[N];
#pragma unroll
        for (int m = 0; m < MT_; ++m)
#pragma unroll
            for (int r = 0; r < 8; ++r) {
                float h = c[m][r] + bias;
                hid[m * 16 + mhi + r][N] = (__bf16)(h > 0.f ? h : 0.f);
            }
    }
    __syncthreads();

    // ---- effect MLP layer 2 + scatter-add (L2-resident atomics) ----
#pragma unroll 1
    for (int t = 0; t < 2; ++t) {
        int nt = wave * 2 + t;
        v8f c[MT_];
#pragma unroll
        for (int m = 0; m < MT_; ++m) c[m] = (v8f){0.f,0.f,0.f,0.f,0.f,0.f,0.f,0.f};
        gemm_tile_m4(&hid[0][0], H_, 8, Wf2p, nt, 8, lane, c);
        int N = nt * 16 + nc;
        float bias = bf2[N];
#pragma unroll
        for (int m = 0; m < MT_; ++m)
#pragma unroll
            for (int r = 0; r < 8; ++r) {
                int ridx = m * 16 + mhi + r;
                int e    = e0 + ridx;
                if (e < E)
                    atomicAdd(&agg[(size_t)s_recv[ridx] * D_ + N], c[m][r] + bias);
            }
    }
}

// ---------------------------------------------------------------------------
// Node kernel: one block = 64 nodes.
//   x = [nodes | agg] ; dn = relu(x @ Wn1 + bn1) @ Wn2 + bn2 ; out = nodes + dn*act
// ---------------------------------------------------------------------------
__global__ __launch_bounds__(128) void ign_node_kernel(
    const float* __restrict__ nodes, const float* __restrict__ agg,
    const float* __restrict__ active_nodes,
    const __bf16* __restrict__ Wn1p, const float* __restrict__ bn1,
    const __bf16* __restrict__ Wn2p, const float* __restrict__ bn2,
    float* __restrict__ nodes_out, int NN)
{
    __shared__ __align__(16) __bf16 xn[64][2 * D_];  // 32 KB
    __shared__ __align__(16) __bf16 hd[64][H_];      // 32 KB

    const int tid  = threadIdx.x;
    const int lane = tid & 31;
    const int wave = tid >> 5;
    const int n0   = blockIdx.x * 64;

#pragma unroll 4
    for (int i = tid; i < 64 * 32; i += 128) {
        const int row = i >> 5;
        const int c4  = (i & 31) << 2;
        int nd = n0 + row; if (nd >= NN) nd = NN - 1;
        const float4 vn = *(const float4*)(nodes + (size_t)nd * D_ + c4);
        const float4 vg = *(const float4*)(agg   + (size_t)nd * D_ + c4);
        cvt_store4(&xn[row][c4],       vn);
        cvt_store4(&xn[row][128 + c4], vg);
    }
    __syncthreads();

    const int mhi = (lane & 16) ? 8 : 0;
    const int nc  = lane & 15;

#pragma unroll 1
    for (int t = 0; t < 4; ++t) {
        int nt = wave * 4 + t;
        v8f c[MT_];
#pragma unroll
        for (int m = 0; m < MT_; ++m) c[m] = (v8f){0.f,0.f,0.f,0.f,0.f,0.f,0.f,0.f};
        gemm_tile_m4(&xn[0][0], 2 * D_, 8, Wn1p, nt, 16, lane, c);
        int N = nt * 16 + nc;
        float bias = bn1[N];
#pragma unroll
        for (int m = 0; m < MT_; ++m)
#pragma unroll
            for (int r = 0; r < 8; ++r) {
                float h = c[m][r] + bias;
                hd[m * 16 + mhi + r][N] = (__bf16)(h > 0.f ? h : 0.f);
            }
    }
    __syncthreads();

#pragma unroll 1
    for (int t = 0; t < 2; ++t) {
        int nt = wave * 2 + t;
        v8f c[MT_];
#pragma unroll
        for (int m = 0; m < MT_; ++m) c[m] = (v8f){0.f,0.f,0.f,0.f,0.f,0.f,0.f,0.f};
        gemm_tile_m4(&hd[0][0], H_, 8, Wn2p, nt, 8, lane, c);
        int N = nt * 16 + nc;
        float bias = bn2[N];
#pragma unroll
        for (int m = 0; m < MT_; ++m)
#pragma unroll
            for (int r = 0; r < 8; ++r) {
                int nd = n0 + m * 16 + mhi + r;
                if (nd < NN) {
                    float dn = (c[m][r] + bias) * active_nodes[nd];
                    nodes_out[(size_t)nd * D_ + N] = nodes[(size_t)nd * D_ + N] + dn;
                }
            }
    }
}

// ---------------------------------------------------------------------------
extern "C" void kernel_launch(void* const* d_in, const int* in_sizes, int n_in,
                              void* d_out, int out_size, void* d_ws, size_t ws_size,
                              hipStream_t stream)
{
    const float* nodes        = (const float*)d_in[0];
    const float* edges        = (const float*)d_in[1];
    const int*   senders      = (const int*)  d_in[2];
    const int*   receivers    = (const int*)  d_in[3];
    const float* active_nodes = (const float*)d_in[4];
    const float* active_edges = (const float*)d_in[5];
    const float* We1 = (const float*)d_in[6];  const float* be1 = (const float*)d_in[7];
    const float* We2 = (const float*)d_in[8];  const float* be2 = (const float*)d_in[9];
    const float* Wf1 = (const float*)d_in[10]; const float* bf1 = (const float*)d_in[11];
    const float* Wf2 = (const float*)d_in[12]; const float* bf2 = (const float*)d_in[13];
    const float* Wn1 = (const float*)d_in[14]; const float* bn1 = (const float*)d_in[15];
    const float* Wn2 = (const float*)d_in[16]; const float* bn2 = (const float*)d_in[17];

    const int NN = in_sizes[4];   // 50000
    const int E  = in_sizes[5];   // 800000

    // workspace layout: fp32 agg [N, D], then bf16 packed weights
    char*  ws  = (char*)d_ws;
    float* agg = (float*)ws;
    size_t off = (size_t)NN * D_ * sizeof(float);
    off = (off + 255) & ~(size_t)255;
    __bf16* We1p = (__bf16*)(ws + off); off += (size_t)XIN_ * H_  * 2;
    __bf16* We2p = (__bf16*)(ws + off); off += (size_t)H_   * DE_ * 2;
    __bf16* Wf1p = (__bf16*)(ws + off); off += (size_t)XIN_ * H_  * 2;
    __bf16* Wf2p = (__bf16*)(ws + off); off += (size_t)H_   * D_  * 2;
    __bf16* Wn1p = (__bf16*)(ws + off); off += (size_t)(2*D_) * H_ * 2;
    __bf16* Wn2p = (__bf16*)(ws + off); off += (size_t)H_   * D_  * 2;

    float* out_nodes = (float*)d_out;
    float* out_edges = out_nodes + (size_t)NN * D_;

    int naggf = NN * D_;
    zero_f32_kernel<<<(naggf + 255) / 256, 256, 0, stream>>>(agg, naggf);

    pack_w_kernel<<<(XIN_ * H_  + 255) / 256, 256, 0, stream>>>(We1, We1p, XIN_, H_);
    pack_w_kernel<<<(H_  * DE_  + 255) / 256, 256, 0, stream>>>(We2, We2p, H_, DE_);
    pack_w_kernel<<<(XIN_ * H_  + 255) / 256, 256, 0, stream>>>(Wf1, Wf1p, XIN_, H_);
    pack_w_kernel<<<(H_  * D_   + 255) / 256, 256, 0, stream>>>(Wf2, Wf2p, H_, D_);
    pack_w_kernel<<<(2*D_ * H_  + 255) / 256, 256, 0, stream>>>(Wn1, Wn1p, 2*D_, H_);
    pack_w_kernel<<<(H_  * D_   + 255) / 256, 256, 0, stream>>>(Wn2, Wn2p, H_, D_);

    ign_edge_kernel<<<(E + 63) / 64, 128, 0, stream>>>(
        nodes, edges, senders, receivers, active_edges,
        We1p, be1, We2p, be2, Wf1p, bf1, Wf2p, bf2,
        agg, out_edges, E);

    ign_node_kernel<<<(NN + 63) / 64, 128, 0, stream>>>(
        nodes, agg, active_nodes, Wn1p, bn1, Wn2p, bn2, out_nodes, NN);
}
